// CapsuleLayer_2388001817155
// MI455X (gfx1250) — matrix operand order repaired and checked by reading, not compiled, not run on hardware
//
#include <hip/hip_runtime.h>
#include <hip/hip_bf16.h>

typedef __attribute__((ext_vector_type(16))) __bf16        v16bf;
typedef __attribute__((ext_vector_type(8)))  float         v8f;
typedef __attribute__((ext_vector_type(4)))  unsigned int  v4u;
typedef __attribute__((ext_vector_type(8)))  int           v8i;
typedef __attribute__((ext_vector_type(4)))  int           v4i;

union FragBF {
    uint4          q[2];
    unsigned short s[16];
    v16bf          v;
};
union FragF32 {
    float4 q4[4];
    float  f[16];
};

// Split fp32 into bf16 hi + bf16 lo (RNE both times).
__device__ __forceinline__ void split_bf16(float f, unsigned short& h, unsigned short& l) {
    unsigned int u  = __float_as_uint(f);
    unsigned int hb = (u + 0x7fffu + ((u >> 16) & 1u)) >> 16;
    h = (unsigned short)hb;
    float hf = __uint_as_float(hb << 16);
    float r  = f - hf;
    unsigned int ur = __float_as_uint(r);
    unsigned int lb = (ur + 0x7fffu + ((ur >> 16) & 1u)) >> 16;
    l = (unsigned short)lb;
}

#define KC      32      // K per chunk (one bf16 WMMA K)
#define NQ      160     // N
#define BROW    40      // padded LDS row stride in ushorts (80 B: TDM pad 4 DW / 16 DW)
#define BBUF    (NQ * BROW)

// ---------------------------------------------------------------------------
// TDM: 2-D tile (32 x 160 bf16, row stride 512 elems) from global 'gsrc'
// (tile origin) into LDS at byte offset 'lds_off', padding 4 DWORDs after
// every 16 DWORDs (row stride 64B -> 80B in LDS, bank-conflict-free frags).
// ---------------------------------------------------------------------------
__device__ __forceinline__ void tdm_stage_b(unsigned lds_off, const unsigned short* gsrc) {
    unsigned long long ga = (unsigned long long)(const void*)gsrc;
    v4u g0;
    g0[0] = 1u;                                    // count=1, user mode, no gather
    g0[1] = lds_off;                               // lds_addr [63:32]
    g0[2] = (unsigned)(ga & 0xffffffffu);          // global_addr [95:64]
    g0[3] = (unsigned)((ga >> 32) & 0x01ffffffu)   // global_addr [120:96]
          | (2u << 30);                            // type = 2 ("image")
    v8i g1;
    g1[0] = (int)((1u << 16)                       // data_size = 1 (2 bytes)
          | (1u << 20)                             // pad_enable
          | (3u << 22)                             // pad_interval: 16 DWORDs
          | (3u << 25));                           // pad_amount: 4 DWORDs
    g1[1] = (int)((512u & 0xffffu) << 16);         // tensor_dim0[15:0] -> bits 63:48
    g1[2] = (int)((512u >> 16) | ((160u & 0xffffu) << 16));   // dim0 hi | dim1 lo
    g1[3] = (int)((160u >> 16) | (32u << 16));     // dim1 hi | tile_dim0 = 32
    g1[4] = (int)(160u);                           // tile_dim1 = 160, tile_dim2 = 0
    g1[5] = (int)512;                              // tensor_dim0_stride[31:0]
    g1[6] = 0;                                     // stride0 hi, stride1 lo
    g1[7] = 0;
    v4i z4; z4[0] = 0; z4[1] = 0; z4[2] = 0; z4[3] = 0;
    v8i z8; z8[0] = 0; z8[1] = 0; z8[2] = 0; z8[3] = 0;
    z8[4] = 0; z8[5] = 0; z8[6] = 0; z8[7] = 0;
    // amdgpu-toolchain (clang-23) 6-arg form
    __builtin_amdgcn_tensor_load_to_lds(g0, g1, z4, z4, z8, 0);
}

__device__ __forceinline__ void tdm_wait_b() {
    __builtin_amdgcn_s_wait_tensorcnt(0);
}

// ---------------------------------------------------------------------------
// Kernel 0: one-time W prep:  W[512,160] f32 -> Wt_hi/Wt_lo [160,512] bf16
// ---------------------------------------------------------------------------
__global__ __launch_bounds__(256) void caps_prep_w(
    const float* __restrict__ W,
    unsigned short* __restrict__ wt_hi, unsigned short* __restrict__ wt_lo)
{
    int idx = blockIdx.x * 256 + threadIdx.x;      // 512*160 = 81920
    if (idx < 81920) {
        int k = idx / 160;
        int n = idx - k * 160;
        unsigned short h, l;
        split_bf16(W[idx], h, l);
        wt_hi[n * 512 + k] = h;
        wt_lo[n * 512 + k] = l;
    }
}

// ---------------------------------------------------------------------------
// Kernel 1: u_hat[147456,160] = x[147456,512] @ W, bf16x3 WMMA.
// 1152 blocks x 256 threads (8 waves). Wave w: rows [w*16,w*16+16), 10 N-tiles.
// A: per-lane direct global loads, register double-buffered.
// B: TDM 2-D tile DMA into double-buffered LDS, wave0-issued, 1 barrier/chunk.
// ---------------------------------------------------------------------------
__global__ __launch_bounds__(256) void caps_gemm_bf16x3(
    const float* __restrict__ x,
    const unsigned short* __restrict__ wt_hi,
    const unsigned short* __restrict__ wt_lo,
    float* __restrict__ uhat)
{
    __shared__ __align__(16) unsigned short bth[2][BBUF];   // 2 x 12.8 KB
    __shared__ __align__(16) unsigned short btl[2][BBUF];   // 2 x 12.8 KB

    const int  t     = threadIdx.x;
    const int  wave  = t >> 5;
    const int  lane  = t & 31;
    const int  hlane = lane & 15;
    const int  hi16  = (lane >> 4) & 1;
    const int  kb    = hi16 ? 8 : 0;
    const long m0    = (long)blockIdx.x * 128;

    const float* arow = x + (m0 + wave * 16 + hlane) * 512;   // this lane's A row

    v8f acc[10];
#pragma unroll
    for (int i = 0; i < 10; ++i)
#pragma unroll
        for (int j = 0; j < 8; ++j) acc[i][j] = 0.0f;

    // ---- preload: B chunk 0 via TDM (wave 0), A chunk 0 into registers
    if (wave == 0) {
        tdm_stage_b((unsigned)(unsigned long long)(const void*)&bth[0][0], wt_hi + 0);
        tdm_stage_b((unsigned)(unsigned long long)(const void*)&btl[0][0], wt_lo + 0);
    }
    FragF32 af[2];
#pragma unroll
    for (int q = 0; q < 2; ++q) {
        af[0].q4[q]     = *(const float4*)(arow + 0 + kb + q * 4);
        af[0].q4[2 + q] = *(const float4*)(arow + 0 + 16 + kb + q * 4);
    }

    for (int c = 0; c < 16; ++c) {          // 16 chunks of K=32
        const int cur = c & 1;

        if (wave == 0) tdm_wait_b();        // B(c) landed in LDS
        __syncthreads();                    // publish B(c); everyone past chunk c-1

        if (c < 15) {
            if (wave == 0) {                // kick B(c+1) DMA under compute
                tdm_stage_b((unsigned)(unsigned long long)(const void*)&bth[1 - cur][0],
                            wt_hi + (c + 1) * KC);
                tdm_stage_b((unsigned)(unsigned long long)(const void*)&btl[1 - cur][0],
                            wt_lo + (c + 1) * KC);
            }
            const float* ap = arow + (c + 1) * KC;    // prefetch A(c+1)
#pragma unroll
            for (int q = 0; q < 2; ++q) {
                af[1 - cur].q4[q]     = *(const float4*)(ap + kb + q * 4);
                af[1 - cur].q4[2 + q] = *(const float4*)(ap + 16 + kb + q * 4);
            }
        }

        // convert A(c) to split-bf16 fragments (ISA 16-bit A 16x32 layout)
        FragBF ahf, alf;
#pragma unroll
        for (int i = 0; i < 16; ++i) split_bf16(af[cur].f[i], ahf.s[i], alf.s[i]);

#pragma unroll
        for (int nt = 0; nt < 10; ++nt) {
            // B fragments: lane = col lane%16, K = 8*(lane>=16)+i (i<8), 16+8*(lane>=16)+i-8 (i>=8), padded rows
            FragBF bhf, blf;
            const unsigned short* bp  = &bth[cur][0] + (nt * 16 + hlane) * BROW + hi16 * 8;
            bhf.q[0] = *(const uint4*)(bp);
            bhf.q[1] = *(const uint4*)(bp + 16);
            const unsigned short* blp = &btl[cur][0] + (nt * 16 + hlane) * BROW + hi16 * 8;
            blf.q[0] = *(const uint4*)(blp);
            blf.q[1] = *(const uint4*)(blp + 16);

            acc[nt] = __builtin_amdgcn_wmma_f32_16x16x32_bf16(
                false, ahf.v, false, bhf.v, (short)0, acc[nt], false, false);
            acc[nt] = __builtin_amdgcn_wmma_f32_16x16x32_bf16(
                false, ahf.v, false, blf.v, (short)0, acc[nt], false, false);
            acc[nt] = __builtin_amdgcn_wmma_f32_16x16x32_bf16(
                false, alf.v, false, bhf.v, (short)0, acc[nt], false, false);
        }
        // single barrier per chunk: next iteration's top barrier separates
        // this compute from the buffer overwrite of chunk c+2.
    }

    // store u_hat; C/D layout: VGPR r -> row r + 8*(lane>=16), col lane%16
    const long mbase = m0 + wave * 16 + (hi16 ? 8 : 0);
#pragma unroll
    for (int nt = 0; nt < 10; ++nt)
#pragma unroll
        for (int r = 0; r < 8; ++r)
            uhat[(mbase + r) * NQ + nt * 16 + hlane] = acc[nt][r];
}

// ---------------------------------------------------------------------------
// Kernel 2: dynamic routing, one block per batch (128 blocks x 512 threads).
// ---------------------------------------------------------------------------
__global__ __launch_bounds__(512) void caps_routing(
    const float* __restrict__ uhat, float* __restrict__ vout)
{
    __shared__ float blog[1152 * 10];
    __shared__ float rmax[1152];
    __shared__ float rinv[1152];
    __shared__ float sv[160];
    __shared__ float vv[160];

    const int  t    = threadIdx.x;
    const long base = (long)blockIdx.x * (1152L * 160L);

    for (int i = t; i < 11520; i += 512) blog[i] = 0.0f;
    __syncthreads();

    for (int it = 0; it < 3; ++it) {
        for (int n = t; n < 1152; n += 512) {
            float mx = blog[n * 10];
#pragma unroll
            for (int c = 1; c < 10; ++c) mx = fmaxf(mx, blog[n * 10 + c]);
            float sum = 0.0f;
#pragma unroll
            for (int c = 0; c < 10; ++c) sum += __expf(blog[n * 10 + c] - mx);
            rmax[n] = mx;
            rinv[n] = 1.0f / sum;
        }
        if (t < 160) sv[t] = 0.0f;
        __syncthreads();

        if (t < 480) {
            int   cd = t % 160;
            int   g  = t / 160;
            int   c  = cd >> 4;
            float part = 0.0f;
            for (int n = g; n < 1152; n += 3) {
                float cc = __expf(blog[n * 10 + c] - rmax[n]) * rinv[n];
                part += cc * uhat[base + (long)n * 160 + cd];
            }
            atomicAdd(&sv[cd], part);
        }
        __syncthreads();

        if (t < 10) {
            float n2 = 0.0f;
#pragma unroll
            for (int d = 0; d < 16; ++d) { float s = sv[t * 16 + d]; n2 += s * s; }
            float nrm = sqrtf(n2) + 1e-7f;
            float fac = (nrm * nrm) / (1.0f + nrm * nrm) / nrm;
#pragma unroll
            for (int d = 0; d < 16; ++d) vv[t * 16 + d] = fac * sv[t * 16 + d];
        }
        __syncthreads();

        if (it < 2) {
            for (int p = t; p < 11520; p += 512) {
                int n = p / 10;
                int c = p - n * 10;
                const float* up = uhat + base + (long)n * 160 + c * 16;
                float dot = 0.0f;
#pragma unroll
                for (int d = 0; d < 16; ++d) dot += up[d] * vv[c * 16 + d];
                blog[p] += dot;
            }
            __syncthreads();
        }
    }

    if (t < 160) vout[(long)blockIdx.x * 160 + t] = vv[t];
}

// ---------------------------------------------------------------------------
extern "C" void kernel_launch(void* const* d_in, const int* in_sizes, int n_in,
                              void* d_out, int out_size, void* d_ws, size_t ws_size,
                              hipStream_t stream) {
    (void)in_sizes; (void)n_in; (void)out_size; (void)ws_size;
    const float* x = (const float*)d_in[0];   // [128,1152,512] f32
    const float* W = (const float*)d_in[1];   // [512,160] f32

    const size_t uhatBytes = 147456ull * 160ull * 4ull;    // 94,371,840
    const size_t wtBytes   = 160ull * 512ull * 2ull;       // 163,840
    float*          uhat  = (float*)d_ws;
    unsigned short* wt_hi = (unsigned short*)((char*)d_ws + uhatBytes);
    unsigned short* wt_lo = (unsigned short*)((char*)d_ws + uhatBytes + wtBytes);
    float* v = (float*)d_out;                 // [128,10,16] f32

    caps_prep_w     <<<dim3(320),  dim3(256), 0, stream>>>(W, wt_hi, wt_lo);
    caps_gemm_bf16x3<<<dim3(1152), dim3(256), 0, stream>>>(x, wt_hi, wt_lo, uhat);
    caps_routing    <<<dim3(128),  dim3(512), 0, stream>>>(uhat, v);
}
